// LSTMDecoder_25838523253400
// MI455X (gfx1250) — matrix operand order repaired and checked by reading, not compiled, not run on hardware
//
#include <hip/hip_runtime.h>
#include <hip/hip_bf16.h>

// ---------------- types / constants ----------------
typedef __attribute__((ext_vector_type(16))) __bf16 v16bf;
typedef __attribute__((ext_vector_type(8)))  __bf16 v8bf;
typedef __attribute__((ext_vector_type(8)))  float  v8f;
typedef __attribute__((ext_vector_type(4)))  float  v4f;

#define H_ENC 512
#define H_DEC 256
#define GATES 2048            // 4*H_ENC
#define KX    1280            // H_DEC + H_ENC + H_ENC  (x | h concat K for gates)

__device__ __forceinline__ v16bf cat16(v8bf lo, v8bf hi) {
  return __builtin_shufflevector(lo, hi, 0, 1, 2, 3, 4, 5, 6, 7,
                                 8, 9, 10, 11, 12, 13, 14, 15);
}

// A-operand fragment (16x32 bf16): lane half h holds K = k0+h*8+{0..7} and k0+16+h*8+{0..7}.
// Pure b128 loads from a pre-converted bf16 row (16B-aligned runs of 8).
__device__ __forceinline__ v16bf load_a_bf(const __bf16* __restrict__ rowk0, int half) {
  v8bf lo = *(const v8bf*)(rowk0 + half * 8);
  v8bf hi = *(const v8bf*)(rowk0 + 16 + half * 8);
  return cat16(lo, hi);
}
// B-operand fragment (32x16 bf16): lane half h holds K = k0+h*8+{0..7} and k0+16+h*8+{0..7} of W row n.
__device__ __forceinline__ v16bf load_b_bf(const __bf16* __restrict__ rowk0, int half) {
  v8bf lo = *(const v8bf*)(rowk0 + half * 8);
  v8bf hi = *(const v8bf*)(rowk0 + 16 + half * 8);
  return cat16(lo, hi);
}

// ---------------- generic C = A * W^T (+bias) on WMMA, bf16 operands ----------------
// Each wave computes a 16(M) x 64(N) strip: 4 accumulators, A fragment reused 4x.
// Inner loop is 10 x global_load_b128 + 4 x v_wmma (no conversion VALU).
__global__ void gemm_bf16_wmma(const __bf16* __restrict__ A, int lda,
                               const __bf16* __restrict__ W, int ldw,
                               const float* __restrict__ bias,
                               float* __restrict__ C, int ldc,
                               int M, int Ncols, int K) {
  const int lane    = threadIdx.x & 31;
  const int wave    = blockIdx.x * (blockDim.x >> 5) + (threadIdx.x >> 5);
  const int tilesN4 = Ncols >> 6;
  const int tilesM  = (M + 15) >> 4;
  if (wave >= tilesM * tilesN4) return;                // wave-uniform exit (EXEC all-1s)
  const int tm = wave / tilesN4;
  const int tn = (wave - tm * tilesN4) << 2;
  const int half = lane >> 4;
  const int l15  = lane & 15;
  const __bf16* aptr = A + (size_t)(tm * 16 + l15) * lda;
  const __bf16* w0 = W + (size_t)((tn + 0) * 16 + l15) * ldw;
  const __bf16* w1 = W + (size_t)((tn + 1) * 16 + l15) * ldw;
  const __bf16* w2 = W + (size_t)((tn + 2) * 16 + l15) * ldw;
  const __bf16* w3 = W + (size_t)((tn + 3) * 16 + l15) * ldw;
  v8f acc0 = {}, acc1 = {}, acc2 = {}, acc3 = {};
  for (int k0 = 0; k0 < K; k0 += 32) {
    v16bf af = load_a_bf(aptr + k0, half);
    v16bf b0 = load_b_bf(w0 + k0, half);
    v16bf b1 = load_b_bf(w1 + k0, half);
    v16bf b2 = load_b_bf(w2 + k0, half);
    v16bf b3 = load_b_bf(w3 + k0, half);
    acc0 = __builtin_amdgcn_wmma_f32_16x16x32_bf16(false, af, false, b0, (short)0, acc0, false, false);
    acc1 = __builtin_amdgcn_wmma_f32_16x16x32_bf16(false, af, false, b1, (short)0, acc1, false, false);
    acc2 = __builtin_amdgcn_wmma_f32_16x16x32_bf16(false, af, false, b2, (short)0, acc2, false, false);
    acc3 = __builtin_amdgcn_wmma_f32_16x16x32_bf16(false, af, false, b3, (short)0, acc3, false, false);
  }
  // C/D layout: reg r -> (M = half*8 + r, N = l15)
#pragma unroll
  for (int j = 0; j < 4; ++j) {
    v8f acc = (j == 0) ? acc0 : (j == 1) ? acc1 : (j == 2) ? acc2 : acc3;
    const int cn = (tn + j) * 16 + l15;
    const float bv = bias ? bias[cn] : 0.0f;
#pragma unroll
    for (int r = 0; r < 8; ++r) {
      int cm = tm * 16 + half * 8 + r;
      C[(size_t)cm * ldc + cn] = acc[r] + bv;
    }
  }
}

// ---------------- gates = [emb(cur) | attended | h] @ [W_ih | W_hh]^T + b_ih + b_hh ----
__device__ __forceinline__ const __bf16* xrow_ptr(const int* __restrict__ cur,
                                                  const __bf16* __restrict__ emb,
                                                  const __bf16* __restrict__ att,
                                                  const __bf16* __restrict__ h,
                                                  int row, int kb) {
  if (kb < H_DEC)         return emb + (size_t)cur[row] * H_DEC + kb;
  if (kb < H_DEC + H_ENC) return att + (size_t)row * H_ENC + (kb - H_DEC);
  return h + (size_t)row * H_ENC + (kb - (H_DEC + H_ENC));
}
__device__ __forceinline__ const __bf16* wrow_ptr(const __bf16* __restrict__ Wih,
                                                  const __bf16* __restrict__ Whh,
                                                  int j, int kb) {
  if (kb < H_DEC + H_ENC) return Wih + (size_t)j * (H_DEC + H_ENC) + kb;
  return Whh + (size_t)j * H_ENC + (kb - (H_DEC + H_ENC));
}
__device__ __forceinline__ v8bf ld8(const __bf16* p) { return *(const v8bf*)p; }

__global__ void gates_gemm_wmma(const int* __restrict__ cur,
                                const __bf16* __restrict__ emb,
                                const __bf16* __restrict__ att,
                                const __bf16* __restrict__ h,
                                const __bf16* __restrict__ Wih,
                                const __bf16* __restrict__ Whh,
                                const float* __restrict__ bih,
                                const float* __restrict__ bhh,
                                float* __restrict__ gates, int B) {
  const int lane    = threadIdx.x & 31;
  const int wave    = blockIdx.x * (blockDim.x >> 5) + (threadIdx.x >> 5);
  const int tilesN4 = GATES >> 6;                      // 32
  const int tilesM  = B >> 4;                          // 8
  if (wave >= tilesM * tilesN4) return;
  const int tm = wave / tilesN4;
  const int tn = (wave - tm * tilesN4) << 2;
  const int half = lane >> 4;
  const int l15  = lane & 15;
  const int arow = tm * 16 + l15;
  const int w0 = (tn + 0) * 16 + l15;
  const int w1 = (tn + 1) * 16 + l15;
  const int w2 = (tn + 2) * 16 + l15;
  const int w3 = (tn + 3) * 16 + l15;
  v8f acc0 = {}, acc1 = {}, acc2 = {}, acc3 = {};
  for (int k0 = 0; k0 < KX; k0 += 32) {
    // 8/16-element runs never cross the 256/768 source boundaries
    v16bf af = cat16(ld8(xrow_ptr(cur, emb, att, h, arow, k0 + half * 8)),
                     ld8(xrow_ptr(cur, emb, att, h, arow, k0 + 16 + half * 8)));
    const int kb = k0 + half * 8;
    v16bf b0 = cat16(ld8(wrow_ptr(Wih, Whh, w0, kb)), ld8(wrow_ptr(Wih, Whh, w0, kb + 16)));
    v16bf b1 = cat16(ld8(wrow_ptr(Wih, Whh, w1, kb)), ld8(wrow_ptr(Wih, Whh, w1, kb + 16)));
    v16bf b2 = cat16(ld8(wrow_ptr(Wih, Whh, w2, kb)), ld8(wrow_ptr(Wih, Whh, w2, kb + 16)));
    v16bf b3 = cat16(ld8(wrow_ptr(Wih, Whh, w3, kb)), ld8(wrow_ptr(Wih, Whh, w3, kb + 16)));
    acc0 = __builtin_amdgcn_wmma_f32_16x16x32_bf16(false, af, false, b0, (short)0, acc0, false, false);
    acc1 = __builtin_amdgcn_wmma_f32_16x16x32_bf16(false, af, false, b1, (short)0, acc1, false, false);
    acc2 = __builtin_amdgcn_wmma_f32_16x16x32_bf16(false, af, false, b2, (short)0, acc2, false, false);
    acc3 = __builtin_amdgcn_wmma_f32_16x16x32_bf16(false, af, false, b3, (short)0, acc3, false, false);
  }
#pragma unroll
  for (int j = 0; j < 4; ++j) {
    v8f acc = (j == 0) ? acc0 : (j == 1) ? acc1 : (j == 2) ? acc2 : acc3;
    const int cn = (tn + j) * 16 + l15;
    const float bv = bih[cn] + bhh[cn];
#pragma unroll
    for (int r = 0; r < 8; ++r) {
      int cm = tm * 16 + half * 8 + r;
      gates[(size_t)cm * GATES + cn] = acc[r] + bv;
    }
  }
}

// ---------------- ragged attention: score/tanh/softmax/weighted segment sum --------
__global__ void attention_kernel(const float* __restrict__ P,
                                 const float* __restrict__ Q,
                                 const float* __restrict__ nhs,
                                 const float* __restrict__ v_att,
                                 const int*  __restrict__ roots,
                                 __bf16* __restrict__ attended_bf,
                                 int Ntot, int B) {
  __shared__ float sc[2048];
  __shared__ float red[256];
  const int b = blockIdx.x;
  const int start = roots[b];
  const int end = (b == B - 1) ? Ntot : roots[b + 1];
  int cnt = end - start; if (cnt > 2048) cnt = 2048;
  const int tid = threadIdx.x;
  const int lane = tid & 31, wid = tid >> 5;
  const int nwaves = blockDim.x >> 5;
  const float* q = Q + (size_t)b * H_ENC;

  // scores: one node per wave, lanes split the H_ENC dot with tanh
  for (int i = wid; i < cnt; i += nwaves) {
    const float* p = P + (size_t)(start + i) * H_ENC;
    float s = 0.f;
    for (int j = lane; j < H_ENC; j += 32)
      s += tanhf(p[j] + q[j]) * v_att[j];
#pragma unroll
    for (int off = 16; off > 0; off >>= 1) s += __shfl_xor(s, off, 32);
    if (lane == 0) sc[i] = s;
  }
  __syncthreads();

  // block max
  float m = -3.4e38f;
  for (int i = tid; i < cnt; i += blockDim.x) m = fmaxf(m, sc[i]);
  red[tid] = m; __syncthreads();
  for (int s2 = 128; s2 > 0; s2 >>= 1) {
    if (tid < s2) red[tid] = fmaxf(red[tid], red[tid + s2]);
    __syncthreads();
  }
  m = red[0]; __syncthreads();

  // exp + sum
  float zs = 0.f;
  for (int i = tid; i < cnt; i += blockDim.x) {
    float e = expf(sc[i] - m); sc[i] = e; zs += e;
  }
  red[tid] = zs; __syncthreads();
  for (int s2 = 128; s2 > 0; s2 >>= 1) {
    if (tid < s2) red[tid] += red[tid + s2];
    __syncthreads();
  }
  const float zinv = 1.f / red[0]; __syncthreads();

  // attended[b][j] = (1/z) * sum_i e_i * nhs[start+i][j]  (bf16 for the gates GEMM)
  for (int j = tid; j < H_ENC; j += blockDim.x) {
    float a = 0.f;
    for (int i = 0; i < cnt; ++i)
      a += sc[i] * nhs[(size_t)(start + i) * H_ENC + j];
    attended_bf[(size_t)b * H_ENC + j] = (__bf16)(a * zinv);
  }
}

// ---------------- LSTM pointwise (PyTorch gate order i,f,g,o) ----------------
__device__ __forceinline__ float sigmoidf(float x) { return 1.f / (1.f + expf(-x)); }

__global__ void lstm_cell_kernel(const float* __restrict__ gates,
                                 float* __restrict__ h, float* __restrict__ c,
                                 __bf16* __restrict__ hbf, int B) {
  int idx = blockIdx.x * blockDim.x + threadIdx.x;
  if (idx >= B * H_ENC) return;
  int b = idx >> 9, u = idx & (H_ENC - 1);
  const float* g = gates + (size_t)b * GATES;
  float ig = sigmoidf(g[u]);
  float fg = sigmoidf(g[u + H_ENC]);
  float gg = tanhf(g[u + 2 * H_ENC]);
  float og = sigmoidf(g[u + 3 * H_ENC]);
  float cn = fg * c[idx] + ig * gg;
  float hn = og * tanhf(cn);
  c[idx] = cn;
  h[idx] = hn;
  hbf[idx] = (__bf16)hn;
}

// ---------------- argmax feedback (first-max tie-break) ----------------
__global__ void argmax_kernel(const float* __restrict__ logits, int* __restrict__ cur, int V) {
  __shared__ float bv[256]; __shared__ int bi[256];
  const int b = blockIdx.x, tid = threadIdx.x;
  const float* row = logits + (size_t)b * V;
  float best = -3.4e38f; int bidx = 0;
  for (int j = tid; j < V; j += blockDim.x) {
    float v = row[j];
    if (v > best) { best = v; bidx = j; }
  }
  bv[tid] = best; bi[tid] = bidx; __syncthreads();
  for (int s = 128; s > 0; s >>= 1) {
    if (tid < s) {
      if (bv[tid + s] > bv[tid] || (bv[tid + s] == bv[tid] && bi[tid + s] < bi[tid])) {
        bv[tid] = bv[tid + s]; bi[tid] = bi[tid + s];
      }
    }
    __syncthreads();
  }
  if (tid == 0) cur[b] = bi[0];
}

// ---------------- init / zero / convert ----------------
__global__ void init_kernel(const float* __restrict__ nhs, const int* __restrict__ roots,
                            const int* __restrict__ labels, float* __restrict__ h,
                            float* __restrict__ c, __bf16* __restrict__ hbf,
                            int* __restrict__ cur, int B) {
  int idx = blockIdx.x * blockDim.x + threadIdx.x;
  if (idx < B * H_ENC) {
    int b = idx >> 9, u = idx & (H_ENC - 1);
    float v = nhs[(size_t)roots[b] * H_ENC + u];
    h[idx] = v;
    hbf[idx] = (__bf16)v;
    c[idx] = 0.f;
  }
  if (idx < B) cur[idx] = labels[idx];   // labels row 0
}

__global__ void zero_kernel(float* __restrict__ p, int n) {
  int i = blockIdx.x * blockDim.x + threadIdx.x;
  if (i < n) p[i] = 0.f;
}

// one-time f32 -> bf16 weight conversion (vectorized 8-wide: b128 in, b128 out, packed cvt)
__global__ void cvt_f32_bf16_kernel(const float* __restrict__ in,
                                    __bf16* __restrict__ out, int n8) {
  int i = blockIdx.x * blockDim.x + threadIdx.x;
  if (i >= n8) return;
  v4f a0 = *(const v4f*)(in + (size_t)i * 8);
  v4f a1 = *(const v4f*)(in + (size_t)i * 8 + 4);
  v8f f = __builtin_shufflevector(a0, a1, 0, 1, 2, 3, 4, 5, 6, 7);
  *(v8bf*)(out + (size_t)i * 8) = __builtin_convertvector(f, v8bf);
}

// ---------------- host driver ----------------
extern "C" void kernel_launch(void* const* d_in, const int* in_sizes, int n_in,
                              void* d_out, int out_size, void* d_ws, size_t ws_size,
                              hipStream_t stream) {
  const float* nhs    = (const float*)d_in[0];
  const int*   labels = (const int*)  d_in[1];
  const int*   roots  = (const int*)  d_in[2];
  const float* emb    = (const float*)d_in[3];
  const float* W_att  = (const float*)d_in[4];
  const float* b_att  = (const float*)d_in[5];
  const float* v_att  = (const float*)d_in[6];
  const float* W_ih   = (const float*)d_in[7];
  const float* W_hh   = (const float*)d_in[8];
  const float* b_ih   = (const float*)d_in[9];
  const float* b_hh   = (const float*)d_in[10];
  const float* W_out  = (const float*)d_in[11];
  const float* b_out  = (const float*)d_in[12];

  const int N  = in_sizes[0] / H_ENC;   // 16384
  const int B  = in_sizes[2];           // 128
  const int L  = in_sizes[1] / B;       // 13
  const int V  = in_sizes[12];          // 8192
  const int VO = in_sizes[3] / H_DEC;   // vocab rows of emb table

  char* ws = (char*)d_ws;
  float*  P      = (float*)ws;  ws += (size_t)N * H_ENC * 4;       // step-invariant attention term
  __bf16* nhsbf  = (__bf16*)ws; ws += (size_t)N * H_ENC * 2;
  __bf16* Wattbf = (__bf16*)ws; ws += (size_t)H_ENC * 2 * H_ENC * 2;
  __bf16* Wihbf  = (__bf16*)ws; ws += (size_t)GATES * (H_DEC + H_ENC) * 2;
  __bf16* Whhbf  = (__bf16*)ws; ws += (size_t)GATES * H_ENC * 2;
  __bf16* Woutbf = (__bf16*)ws; ws += (size_t)V * H_ENC * 2;
  __bf16* embbf  = (__bf16*)ws; ws += (size_t)VO * H_DEC * 2;
  float*  Q      = (float*)ws;  ws += (size_t)B * H_ENC * 4;
  float*  h      = (float*)ws;  ws += (size_t)B * H_ENC * 4;
  float*  c      = (float*)ws;  ws += (size_t)B * H_ENC * 4;
  __bf16* hbf    = (__bf16*)ws; ws += (size_t)B * H_ENC * 2;
  __bf16* attbf  = (__bf16*)ws; ws += (size_t)B * H_ENC * 2;
  float*  gate   = (float*)ws;  ws += (size_t)B * GATES * 4;
  int*    cur    = (int*)ws;    ws += (size_t)B * 4;

  float* out = (float*)d_out;

  // one-time f32 -> bf16 conversions (all step-invariant operands)
  auto cvt = [&](const float* src, __bf16* dst, size_t n) {
    int n8 = (int)(n / 8);
    cvt_f32_bf16_kernel<<<(n8 + 255) / 256, 256, 0, stream>>>(src, dst, n8);
  };
  cvt(nhs,   nhsbf,  (size_t)N * H_ENC);
  cvt(W_att, Wattbf, (size_t)H_ENC * 2 * H_ENC);
  cvt(W_ih,  Wihbf,  (size_t)GATES * (H_DEC + H_ENC));
  cvt(W_hh,  Whhbf,  (size_t)GATES * H_ENC);
  cvt(W_out, Woutbf, (size_t)V * H_ENC);
  cvt(emb,   embbf,  (size_t)VO * H_DEC);

  // outputs[0] stays zero (d_out is poisoned by harness)
  zero_kernel<<<(B * V + 255) / 256, 256, 0, stream>>>(out, B * V);
  init_kernel<<<(B * H_ENC + 255) / 256, 256, 0, stream>>>(nhs, roots, labels, h, c, hbf, cur, B);

  // P = nhs @ W_att[:, H_ENC:].T + b_att   (done once; removes 17 GFLOP/step)
  {
    int waves = (N / 16) * (H_ENC / 64);
    gemm_bf16_wmma<<<(waves + 7) / 8, 256, 0, stream>>>(
        nhsbf, H_ENC, Wattbf + H_ENC, 2 * H_ENC, b_att, P, H_ENC, N, H_ENC, H_ENC);
  }

  for (int t = 0; t < L - 1; ++t) {
    { // Q = h @ W_att[:, :H_ENC].T
      int waves = (B / 16) * (H_ENC / 64);
      gemm_bf16_wmma<<<(waves + 7) / 8, 256, 0, stream>>>(
          hbf, H_ENC, Wattbf, 2 * H_ENC, nullptr, Q, H_ENC, B, H_ENC, H_ENC);
    }
    attention_kernel<<<B, 256, 0, stream>>>(P, Q, nhs, v_att, roots, attbf, N, B);
    { // gates = [emb(cur)|att|h] @ [W_ih|W_hh]^T + b_ih + b_hh
      int waves = (B / 16) * (GATES / 64);
      gates_gemm_wmma<<<(waves + 7) / 8, 256, 0, stream>>>(
          cur, embbf, attbf, hbf, Wihbf, Whhbf, b_ih, b_hh, gate, B);
    }
    lstm_cell_kernel<<<(B * H_ENC + 255) / 256, 256, 0, stream>>>(gate, h, c, hbf, B);
    float* logits = out + (size_t)(t + 1) * B * V;
    { // logits = h @ W_out^T + b_out
      int waves = (B / 16) * (V / 64);
      gemm_bf16_wmma<<<(waves + 7) / 8, 256, 0, stream>>>(
          hbf, H_ENC, Woutbf, H_ENC, b_out, logits, V, B, H_ENC, H_ENC);
    }
    argmax_kernel<<<B, 256, 0, stream>>>(logits, cur, V);
  }
}
